// LinearAttention_35665408426496
// MI455X (gfx1250) — matrix operand order repaired and checked by reading, not compiled, not run on hardware
//
#include <hip/hip_runtime.h>
#include <hip/hip_bf16.h>
#include <math.h>

typedef __bf16 bf16_t;
typedef __attribute__((ext_vector_type(16))) __bf16 v16bf;
typedef __attribute__((ext_vector_type(8)))  __bf16 v8bf;
typedef __attribute__((ext_vector_type(4)))  __bf16 v4bf;
typedef __attribute__((ext_vector_type(8)))  float  v8f;

#define BDIM 4
#define TDIM 4096
#define DDIM 1024
#define HNUM 16
#define HD   64
#define MROWS (BDIM * TDIM)   // 16384
#define NCHUNK 8              // T-chunks for kv partial reduction
#define TCHUNK (TDIM / NCHUNK)

// ---------------------------------------------------------------------------
// Fragment loaders (bf16, row-major source, row stride = DDIM elements)
// A-matrix 16x32 layout (ISA 7.12.2): lanes 0-15 hold K = 0..7 & 16..23,
// lanes 16-31 hold K = 8..15 & 24..31 (2 bf16 per VGPR).
// ---------------------------------------------------------------------------
__device__ __forceinline__ v16bf ld_a_frag(const bf16_t* p) {
  const v8bf lo = *reinterpret_cast<const v8bf*>(p);        // K + 0..7
  const v8bf hi = *reinterpret_cast<const v8bf*>(p + 16);   // K + 16..23
  return __builtin_shufflevector(lo, hi, 0,1,2,3,4,5,6,7,8,9,10,11,12,13,14,15);
}
// B-matrix 32x16 layout (same K order as A): lanes 0-15 hold K = 0..7 & 16..23,
// lanes 16-31 hold K = 8..15 & 24..31.
__device__ __forceinline__ v16bf ld_b_frag(const bf16_t* p) {
  const v8bf lo = *reinterpret_cast<const v8bf*>(p);        // K + 0..7
  const v8bf hi = *reinterpret_cast<const v8bf*>(p + 16);   // K + 16..23
  return __builtin_shufflevector(lo, hi, 0,1,2,3,4,5,6,7,8,9,10,11,12,13,14,15);
}

// ---------------------------------------------------------------------------
// WMMA GEMM:  C[M,N] = A[M,K] * W[N,K]^T (+ bias, + optional elu+1)
// M = MROWS, N = K = DDIM.  Block = 256 threads = 8 waves, block tile 128x128,
// wave tile 32x64 -> 8 v_wmma_f32_16x16x32_bf16 per K-step of 32.
// MODE 0: out = acc + bias   (fp32)
// MODE 1: out = elu(acc + bias) + 1  (fp32)  [= x+1 if x>0 else exp(x)]
// ---------------------------------------------------------------------------
template <int MODE>
__global__ __launch_bounds__(256) void gemm_bf16_wmma(
    const bf16_t* __restrict__ A, const bf16_t* __restrict__ W,
    const float* __restrict__ bias, float* __restrict__ Cmat) {
  const int lane = threadIdx.x & 31;
  const int wave = threadIdx.x >> 5;
  const int wm   = wave & 3;    // 4 wave rows  * 32
  const int wn   = wave >> 2;   // 2 wave cols  * 64
  const int lrow = lane & 15;
  const int half = lane >> 4;

  const int row0 = blockIdx.y * 128 + wm * 32;
  const int col0 = blockIdx.x * 128 + wn * 64;

  const bf16_t* pa[2];
  pa[0] = A + (size_t)(row0 + lrow) * DDIM + half * 8;
  pa[1] = pa[0] + (size_t)16 * DDIM;
  const bf16_t* pw[4];
  pw[0] = W + (size_t)(col0 + lrow) * DDIM + half * 8;
  pw[1] = pw[0] + (size_t)16 * DDIM;
  pw[2] = pw[0] + (size_t)32 * DDIM;
  pw[3] = pw[0] + (size_t)48 * DDIM;

  v8f acc[2][4] = {};

  for (int k0 = 0; k0 < DDIM; k0 += 32) {
    v16bf af[2], wf[4];
    af[0] = ld_a_frag(pa[0] + k0);
    af[1] = ld_a_frag(pa[1] + k0);
#pragma unroll
    for (int i = 0; i < 4; ++i) wf[i] = ld_b_frag(pw[i] + k0);
#pragma unroll
    for (int mi = 0; mi < 2; ++mi)
#pragma unroll
      for (int ni = 0; ni < 4; ++ni)
        acc[mi][ni] = __builtin_amdgcn_wmma_f32_16x16x32_bf16(
            false, af[mi], false, wf[ni], (short)0, acc[mi][ni], false, false);
  }

  // C/D layout: VGPR r -> M = r (lanes 0-15) / M = 8+r (lanes 16-31), N = lrow
#pragma unroll
  for (int mi = 0; mi < 2; ++mi) {
    const int rowb = row0 + mi * 16 + half * 8;
#pragma unroll
    for (int ni = 0; ni < 4; ++ni) {
      const int col = col0 + ni * 16 + lrow;
      const float bb = bias[col];
#pragma unroll
      for (int r = 0; r < 8; ++r) {
        float vv = acc[mi][ni][r] + bb;
        if (MODE == 1) vv = (vv > 0.f) ? (vv + 1.f) : __expf(vv);
        Cmat[(size_t)(rowb + r) * DDIM + col] = vv;
      }
    }
  }
}

// ---------------------------------------------------------------------------
// fp32 -> bf16 conversion, 4 elements / thread
// ---------------------------------------------------------------------------
__global__ __launch_bounds__(256) void cvt_f32_bf16(
    const float* __restrict__ in, bf16_t* __restrict__ out, int n4) {
  const int i = blockIdx.x * 256 + threadIdx.x;
  if (i < n4) {
    const float4 f = reinterpret_cast<const float4*>(in)[i];
    v4bf o;
    o[0] = (bf16_t)f.x; o[1] = (bf16_t)f.y; o[2] = (bf16_t)f.z; o[3] = (bf16_t)f.w;
    reinterpret_cast<v4bf*>(out)[i] = o;
  }
}

// ---------------------------------------------------------------------------
// kv[b,h,d,e] = sum_t k[b,t,h,d]*v[b,t,h,e];  ksum[b,h,d] = sum_t k
// Partial over T-chunks of TCHUNK; grid (B*H, NCHUNK), 256 threads.
// Thread -> (d = tid>>2, 16 e-values at e0=(tid&3)*16).
// ---------------------------------------------------------------------------
__global__ __launch_bounds__(256) void kv_partial(
    const float* __restrict__ kf, const float* __restrict__ vf,
    float* __restrict__ kvp, float* __restrict__ ksp) {
  const int bh = blockIdx.x;            // 0..63
  const int chunk = blockIdx.y;         // 0..NCHUNK-1
  const int b = bh >> 4, h = bh & 15;
  const int tid = threadIdx.x;
  const int d = tid >> 2;
  const int e0 = (tid & 3) * 16;
  __shared__ float sk[8][64];
  __shared__ float sv[8][64];
  float acc[16];
#pragma unroll
  for (int i = 0; i < 16; ++i) acc[i] = 0.f;
  float ks = 0.f;
  const size_t base = ((size_t)b * TDIM + (size_t)chunk * TCHUNK) * DDIM + h * HD;
  for (int t0 = 0; t0 < TCHUNK; t0 += 8) {
#pragma unroll
    for (int j = 0; j < 2; ++j) {
      const int idx = tid + j * 256;          // 0..511
      const int tt = idx >> 6, dd = idx & 63;
      sk[tt][dd] = kf[base + (size_t)(t0 + tt) * DDIM + dd];
      sv[tt][dd] = vf[base + (size_t)(t0 + tt) * DDIM + dd];
    }
    __syncthreads();
#pragma unroll
    for (int tt = 0; tt < 8; ++tt) {
      const float kd = sk[tt][d];
      if ((tid & 3) == 0) ks += kd;
#pragma unroll
      for (int i = 0; i < 16; ++i) acc[i] = fmaf(kd, sv[tt][e0 + i], acc[i]);
    }
    __syncthreads();
  }
  float* dst = kvp + (size_t)(bh * NCHUNK + chunk) * (HD * HD);
#pragma unroll
  for (int i = 0; i < 16; ++i) dst[d * HD + e0 + i] = acc[i];
  if ((tid & 3) == 0) ksp[(size_t)(bh * NCHUNK + chunk) * HD + d] = ks;
}

__global__ __launch_bounds__(256) void kv_reduce(
    const float* __restrict__ kvp, const float* __restrict__ ksp,
    float* __restrict__ kvm, float* __restrict__ ksum) {
  const int i = blockIdx.x * 256 + threadIdx.x;  // 0 .. 64*4096-1
  const int bh = i >> 12, j = i & 4095;
  float s = 0.f;
#pragma unroll
  for (int p = 0; p < NCHUNK; ++p) s += kvp[(size_t)(bh * NCHUNK + p) * 4096 + j];
  kvm[i] = s;
  if (j < HD) {
    float s2 = 0.f;
#pragma unroll
    for (int p = 0; p < NCHUNK; ++p) s2 += ksp[(size_t)(bh * NCHUNK + p) * HD + j];
    ksum[bh * HD + j] = s2;
  }
}

// ---------------------------------------------------------------------------
// attn[b,t,h,e] = (sum_d q[b,t,h,d]*kv[b,h,d,e]) / (sum_d q*ksum + 1e-8)
// Output directly in bf16 for the final WMMA projection.
// Grid (B*H, T/128); kv head tile (16 KB) staged in LDS; 4 t-rows / iter.
// ---------------------------------------------------------------------------
__global__ __launch_bounds__(256) void attn_norm(
    const float* __restrict__ qf, const float* __restrict__ kvm,
    const float* __restrict__ ksum, bf16_t* __restrict__ attn) {
  const int bh = blockIdx.x, b = bh >> 4, h = bh & 15;
  const int t0 = blockIdx.y * 128;
  __shared__ float skv[HD * HD];   // 16 KB
  __shared__ float sks[HD];
  __shared__ float sq[4][HD];
  const int tid = threadIdx.x;
#pragma unroll
  for (int j = 0; j < 16; ++j) skv[tid + j * 256] = kvm[(size_t)bh * 4096 + tid + j * 256];
  if (tid < HD) sks[tid] = ksum[bh * HD + tid];
  __syncthreads();
  const int r = tid >> 6, e = tid & 63;
  for (int tb = 0; tb < 128; tb += 4) {
    sq[r][e] = qf[(size_t)(b * TDIM + t0 + tb + r) * DDIM + h * HD + e];
    __syncthreads();
    float num = 0.f, den = 0.f;
#pragma unroll 8
    for (int d = 0; d < HD; ++d) {
      const float qd = sq[r][d];
      num = fmaf(qd, skv[d * HD + e], num);
      den = fmaf(qd, sks[d], den);
    }
    attn[(size_t)(b * TDIM + t0 + tb + r) * DDIM + h * HD + e] =
        (bf16_t)(num / (den + 1e-8f));
    __syncthreads();
  }
}

// ---------------------------------------------------------------------------
extern "C" void kernel_launch(void* const* d_in, const int* in_sizes, int n_in,
                              void* d_out, int out_size, void* d_ws, size_t ws_size,
                              hipStream_t stream) {
  (void)in_sizes; (void)n_in; (void)out_size; (void)ws_size;
  const float* x  = (const float*)d_in[0];
  const float* Wq = (const float*)d_in[1];
  const float* bq = (const float*)d_in[2];
  const float* Wk = (const float*)d_in[3];
  const float* bk = (const float*)d_in[4];
  const float* Wv = (const float*)d_in[5];
  const float* bv = (const float*)d_in[6];
  const float* Wo = (const float*)d_in[7];
  const float* bo = (const float*)d_in[8];
  float* out = (float*)d_out;

  char* p = (char*)d_ws;
  auto alloc = [&](size_t bytes) -> char* {
    char* r = p;
    p += (bytes + 255) & ~(size_t)255;
    return r;
  };
  bf16_t* xb   = (bf16_t*)alloc((size_t)MROWS * DDIM * 2);
  bf16_t* wqb  = (bf16_t*)alloc((size_t)DDIM * DDIM * 2);
  bf16_t* wkb  = (bf16_t*)alloc((size_t)DDIM * DDIM * 2);
  bf16_t* wvb  = (bf16_t*)alloc((size_t)DDIM * DDIM * 2);
  bf16_t* wob  = (bf16_t*)alloc((size_t)DDIM * DDIM * 2);
  float*  qf   = (float*)alloc((size_t)MROWS * DDIM * 4);
  float*  kf   = (float*)alloc((size_t)MROWS * DDIM * 4);
  float*  vf   = (float*)alloc((size_t)MROWS * DDIM * 4);
  float*  kvp  = (float*)alloc((size_t)64 * NCHUNK * HD * HD * 4);
  float*  ksp  = (float*)alloc((size_t)64 * NCHUNK * HD * 4);
  float*  kvm  = (float*)alloc((size_t)64 * HD * HD * 4);
  float*  ksum = (float*)alloc((size_t)64 * HD * 4);
  bf16_t* attnb = xb;  // x (bf16) is dead after the v projection: reuse

  // fp32 -> bf16
  const int nx4 = MROWS * DDIM / 4;
  const int nw4 = DDIM * DDIM / 4;
  cvt_f32_bf16<<<(nx4 + 255) / 256, 256, 0, stream>>>(x, xb, nx4);
  cvt_f32_bf16<<<(nw4 + 255) / 256, 256, 0, stream>>>(Wq, wqb, nw4);
  cvt_f32_bf16<<<(nw4 + 255) / 256, 256, 0, stream>>>(Wk, wkb, nw4);
  cvt_f32_bf16<<<(nw4 + 255) / 256, 256, 0, stream>>>(Wv, wvb, nw4);
  cvt_f32_bf16<<<(nw4 + 255) / 256, 256, 0, stream>>>(Wo, wob, nw4);

  // q/k/v projections (WMMA), feature map fused on q and k
  const dim3 gg(DDIM / 128, MROWS / 128);  // (8, 128)
  gemm_bf16_wmma<1><<<gg, 256, 0, stream>>>(xb, wqb, bq, qf);
  gemm_bf16_wmma<1><<<gg, 256, 0, stream>>>(xb, wkb, bk, kf);
  gemm_bf16_wmma<0><<<gg, 256, 0, stream>>>(xb, wvb, bv, vf);

  // kv / ksum einsums + normalization
  kv_partial<<<dim3(64, NCHUNK), 256, 0, stream>>>(kf, vf, kvp, ksp);
  kv_reduce<<<(64 * 4096) / 256, 256, 0, stream>>>(kvp, ksp, kvm, ksum);
  attn_norm<<<dim3(64, TDIM / 128), 256, 0, stream>>>(qf, kvm, ksum, attnb);

  // output projection (WMMA) -> fp32 result
  gemm_bf16_wmma<0><<<gg, 256, 0, stream>>>(attnb, wob, bo, out);
}
